// S4_67405216743723
// MI455X (gfx1250) — matrix operand order repaired and checked by reading, not compiled, not run on hardware
//
#include <hip/hip_runtime.h>
#include <math.h>

// ---------------- problem constants (from reference) ----------------
#define B_   8
#define L_   5000
#define DIN  12
#define H_   256
#define N2_  32
#define NLAYERS 6
#define LSH  48            // left zero-pad of kernel so Toeplitz indices >= 0
#define LKP  5120          // conceptual forward kernel row length (mult of 32)
#define LKPR 5136          // reversed kernel row in global, +16 zero tail
#define LKPC 5128          // one shifted LDS copy (LKP + 8), 16B multiple
#define LUP  5024          // padded input row length (mult of 32, >= 32*156+31)
#define LYP  5008          // padded conv-output length = 313*16
#define NC   313           // number of 16-wide output chunks
#define EPSBN 1e-5f

typedef __attribute__((ext_vector_type(16))) __bf16       v16bf;
typedef __attribute__((ext_vector_type(8)))  float        v8f;
typedef __attribute__((ext_vector_type(4)))  unsigned int u32x4;
typedef __attribute__((ext_vector_type(8)))  int          i32x8;
typedef __attribute__((ext_vector_type(4)))  int          i32x4;

union Frag16 { v16bf v; unsigned short u[16]; u32x4 q[2]; };

__device__ __forceinline__ unsigned short f2bf(float f) {
  unsigned u = __builtin_bit_cast(unsigned, f);
  u += 0x7FFFu + ((u >> 16) & 1u);          // round-to-nearest-even
  return (unsigned short)(u >> 16);
}

// ---------------- Tensor Data Mover staging (guarded) ----------------
#if defined(__HIP_DEVICE_COMPILE__) && __has_builtin(__builtin_amdgcn_tensor_load_to_lds)
#define HAVE_TDM 1
#else
#define HAVE_TDM 0
#endif

// 2D tile of 2-byte elements: tile_d1 rows of tile_d0 elems, global row stride
// stride0 (elems), packed contiguously into LDS at byte offset lds_off.
// D# packing per CDNA5 ISA.
__device__ __forceinline__ void tdm_load_2d_b16(unsigned lds_off, const void* gsrc,
                                                unsigned tile_d0, unsigned tile_d1,
                                                unsigned long long stride0) {
#if HAVE_TDM
  unsigned long long ga = (unsigned long long)(size_t)gsrc;
  u32x4 g0;
  g0[0] = 1u;                                              // count=1, user desc
  g0[1] = lds_off;                                         // lds_addr
  g0[2] = (unsigned)ga;                                    // global_addr[31:0]
  g0[3] = (unsigned)((ga >> 32) & 0x01FFFFFFull) | (2u << 30); // addr[56:32], type=2
  unsigned td0 = tile_d0, td1 = tile_d1;                   // tensor dims == tile (no OOB)
  i32x8 g1;
  g1[0] = (int)(1u << 16);                                 // wg_mask=0, data_size=1 (2B)
  g1[1] = (int)((td0 & 0xFFFFu) << 16);                    // tensor_dim0[15:0]
  g1[2] = (int)(((td0 >> 16) & 0xFFFFu) | ((td1 & 0xFFFFu) << 16));
  g1[3] = (int)(((td1 >> 16) & 0xFFFFu) | ((tile_d0 & 0xFFFFu) << 16)); // tile_dim0
  g1[4] = (int)(tile_d1 & 0xFFFFu);                        // tile_dim1, tile_dim2=0
  g1[5] = (int)(unsigned)(stride0 & 0xFFFFFFFFull);        // dim0_stride[31:0]
  g1[6] = (int)(unsigned)((stride0 >> 32) & 0xFFFFull);    // dim0_stride[47:32]
  g1[7] = 0;
  i32x4 z4 = {0, 0, 0, 0};
#if defined(__clang_major__) && (__clang_major__ >= 23)
  i32x8 z8 = {0, 0, 0, 0, 0, 0, 0, 0};
  __builtin_amdgcn_tensor_load_to_lds(g0, g1, z4, z4, z8, 0);
#else
  __builtin_amdgcn_tensor_load_to_lds(g0, g1, z4, z4, 0);
#endif
#else
  (void)lds_off; (void)gsrc; (void)tile_d0; (void)tile_d1; (void)stride0;
#endif
}

__device__ __forceinline__ void wait_tensor0() {
#if defined(__HIP_DEVICE_COMPILE__)
#if __has_builtin(__builtin_amdgcn_s_wait_tensorcnt)
  __builtin_amdgcn_s_wait_tensorcnt(0);
#else
  asm volatile("s_wait_tensorcnt 0x0" ::: "memory");
#endif
#endif
}

// ---------------- small elementwise / setup kernels ----------------
__global__ void preproj_kernel(const float* __restrict__ x, const float* __restrict__ pw,
                               const float* __restrict__ pb, float* __restrict__ hA) {
  size_t idx = (size_t)blockIdx.x * 256 + threadIdx.x;
  if (idx >= (size_t)B_ * H_ * L_) return;
  int l = (int)(idx % L_);
  int h = (int)((idx / L_) % H_);
  int b = (int)(idx / ((size_t)H_ * L_));
  const float* xp = x + ((size_t)b * L_ + l) * DIN;
  const float* wp = pw + (size_t)h * DIN;
  float s = pb[h];
#pragma unroll
  for (int d = 0; d < DIN; ++d) s += xp[d] * wp[d];
  hA[idx] = s;
}

__global__ void wcvt_kernel(const float* __restrict__ w, unsigned short* __restrict__ w16, int n) {
  int i = blockIdx.x * 256 + threadIdx.x;
  if (i < n) w16[i] = f2bf(w[i]);
}

__global__ void ctilde_kernel(const float* __restrict__ log_dt, const float* __restrict__ C_re,
                              const float* __restrict__ C_im, const float* __restrict__ lAr,
                              const float* __restrict__ Aim, float* __restrict__ Ctr,
                              float* __restrict__ Cti, float* __restrict__ dAr,
                              float* __restrict__ dAi) {
  int idx = blockIdx.x * 256 + threadIdx.x;
  if (idx >= H_ * N2_) return;
  int h = idx / N2_;
  float dt  = expf(log_dt[h]);
  float are = -expf(lAr[idx]);
  float aim = Aim[idx];
  float dre = are * dt, dim = aim * dt;
  float ex = expf(dre);
  float er = ex * cosf(dim) - 1.0f;                // exp(dtA) - 1
  float ei = ex * sinf(dim);
  float cr = C_re[idx], ci = C_im[idx];
  float nr = cr * er - ci * ei;                    // C*(exp(dtA)-1)
  float ni = cr * ei + ci * er;
  float den = are * are + aim * aim;               // / A
  Ctr[idx] = (nr * are + ni * aim) / den;
  Cti[idx] = (ni * are - nr * aim) / den;
  dAr[idx] = dre;  dAi[idx] = dim;
}

// Forward kernel value at index t is kx[t] = k[t-LSH] (zero outside [LSH, L+LSH)).
// We store it REVERSED: kxr[M - t] = kx[t], M = LKP-1, so the Toeplitz gather in
// the conv kernel is ascending/contiguous.  Tail [LKP, LKPR) zeroed.
__global__ __launch_bounds__(256) void kbuild_kernel(const float* __restrict__ Ctr,
                                                     const float* __restrict__ Cti,
                                                     const float* __restrict__ dAr,
                                                     const float* __restrict__ dAi,
                                                     unsigned short* __restrict__ kxr16) {
  __shared__ float sC[4][N2_];
  const int h = blockIdx.y;
  const int t = blockIdx.x * 256 + threadIdx.x;
  if (threadIdx.x < N2_) {
    sC[0][threadIdx.x] = Ctr[h * N2_ + threadIdx.x];
    sC[1][threadIdx.x] = Cti[h * N2_ + threadIdx.x];
    sC[2][threadIdx.x] = dAr[h * N2_ + threadIdx.x];
    sC[3][threadIdx.x] = dAi[h * N2_ + threadIdx.x];
  }
  __syncthreads();
  if (blockIdx.x == 0 && threadIdx.x < (LKPR - LKP))       // zero tail
    kxr16[(size_t)h * LKPR + LKP + threadIdx.x] = 0;
  if (t >= LKP) return;
  float val = 0.f;
  const int td = t - LSH;
  if (td >= 0 && td < L_) {
    float acc = 0.f, ft = (float)td;
#pragma unroll 4
    for (int n = 0; n < N2_; ++n) {
      float e  = expf(sC[2][n] * ft);
      float ph = sC[3][n] * ft;
      acc += e * (sC[0][n] * cosf(ph) - sC[1][n] * sinf(ph));
    }
    val = 2.0f * acc;
  }
  kxr16[(size_t)h * LKPR + ((LKP - 1) - t)] = f2bf(val);
}

__global__ void u16_kernel(const float* __restrict__ hA, unsigned short* __restrict__ u16g) {
  size_t idx = (size_t)blockIdx.x * 256 + threadIdx.x;
  if (idx >= (size_t)B_ * H_ * LUP) return;
  int t = (int)(idx % LUP);
  size_t bh = idx / LUP;
  float v = (t < L_) ? hA[bh * L_ + t] : 0.f;
  u16g[idx] = f2bf(v);
}

// ---------------- causal conv as Toeplitz WMMA, one WG per channel ----------------
// y[b,l] = sum_m k[l-m] u[b,m].  A[r,kk] = kx[16i-32j+r-kk+LSH] = kxr[offA+kk] with
// offA = M - (16i+r+LSH) + 32j + kb0.  The parity s = offA & 7 is invariant in i and j,
// so each lane reads from one of 8 element-shifted LDS copies of kxr with two aligned
// ds_load_b128 per fragment (4 DS ops per WMMA total).  All copies + the 8 batch rows
// of u live in LDS (159 KB of the 320 KB WGP pool), staged by 9 TDM descriptors.
// Epilogue fuses the D-skip + exact-erf GELU and writes bf16 activations
// h-contiguous for the following GEMM: ygt[b][l][h].
__global__ __launch_bounds__(256) void conv_wmma_kernel(
    const unsigned short* __restrict__ u16g,    // [B][H][LUP] bf16
    const unsigned short* __restrict__ kxr16,   // [H][LKPR]   bf16 (reversed rows)
    const float* __restrict__ hA,               // [B][H][L]   layer input (skip)
    const float* __restrict__ Dvec,             // [H]
    unsigned short* __restrict__ ygt)           // [B][LYP][H] bf16
{
  extern __shared__ unsigned short smem[];      // u: B_*LUP | 8 copies of LKPC
  unsigned short* uL = smem;
  unsigned short* kC = smem + B_ * LUP;
  const int h    = blockIdx.x;
  const int tid  = threadIdx.x;
  const int wave = tid >> 5;
  const int lane = tid & 31;
  const unsigned short* krow = kxr16 + (size_t)h * LKPR;

#if HAVE_TDM
  if (wave == 0) {
    unsigned lds0 = (unsigned)(size_t)(void*)smem;
    tdm_load_2d_b16(lds0, u16g + (size_t)h * LUP, LUP, B_, (unsigned long long)H_ * LUP);
    unsigned ldsk = lds0 + (unsigned)(B_ * LUP) * 2u;
#pragma unroll
    for (int s = 0; s < 8; ++s)                 // 8 element-shifted copies
      tdm_load_2d_b16(ldsk + (unsigned)(s * LKPC) * 2u, krow + s, LKPC, 1, 0ull);
    wait_tensor0();
  }
  __syncthreads();
#else
  for (int b = 0; b < B_; ++b) {
    const u32x4* src = (const u32x4*)(u16g + ((size_t)b * H_ + h) * LUP);
    u32x4* dst = (u32x4*)(uL + b * LUP);
    for (int i = tid; i < LUP / 8; i += 256) dst[i] = src[i];
  }
  for (int s = 0; s < 8; ++s) {
    unsigned short* dst = kC + s * LKPC;
    for (int t = tid; t < LKPC; t += 256) dst[t] = krow[t + s];
  }
  __syncthreads();
#endif

  const float Dh   = Dvec[h];
  const int r      = lane & 15;                 // A row / B col / C col
  const int kb0A   = (lane < 16) ? 0 : 8;       // A layout: K halves 0-7/16-23 vs 8-15/24-31
  const int kb0B   = (lane < 16) ? 0 : 8;       // B: same K halves as A (0-7/16-23 vs 8-15/24-31)
  const int rowadd = (lane < 16) ? 0 : 8;       // C/D: VGPR v -> M=v or v+8
  const int M      = LKP - 1;
  const int sA     = (M - LSH - r + kb0A) & 7;  // copy selector (i,j invariant)
  const unsigned short* kLane = kC + sA * LKPC - sA;   // kLane[e] == kxr[e], 16B-aligned use

  for (int i = wave; i < NC; i += 8) {
    v8f acc = {0.f, 0.f, 0.f, 0.f, 0.f, 0.f, 0.f, 0.f};
    const int jmax  = (16 * i + 15) >> 5;
    const int offA0 = M - (16 * i + r + LSH) + kb0A;   // (offA0 - sA) % 8 == 0
    const unsigned short* kp = kLane + offA0;
    for (int j = 0; j <= jmax; ++j) {
      Frag16 a, b;
      const u32x4* pA = (const u32x4*)(kp + 32 * j);   // aligned: base has elem%8==0
      a.q[0] = pA[0];                                  // K = kb0A .. +7
      a.q[1] = pA[2];                                  // K = kb0A+16 .. +23
      if (r < B_) {
        const u32x4* pB = (const u32x4*)(uL + r * LUP + 32 * j + kb0B);
        b.q[0] = pB[0];  b.q[1] = pB[2];
      } else {
        u32x4 z = {0, 0, 0, 0};
        b.q[0] = z;  b.q[1] = z;
      }
      acc = __builtin_amdgcn_wmma_f32_16x16x32_bf16(false, a.v, false, b.v,
                                                    (short)0, acc, false, false);
    }
    if (r < B_) {
#pragma unroll
      for (int v = 0; v < 8; ++v) {
        const int l = 16 * i + v + rowadd;      // <= 5007 < LYP
        float y = 0.f;
        if (l < L_) {
          const float uu = hA[((size_t)r * H_ + h) * L_ + l];
          y = acc[v] + Dh * uu;                 // skip connection
          y = 0.5f * y * (1.0f + erff(y * 0.70710678118654752f)); // exact-erf GELU
        }
        ygt[((size_t)r * LYP + l) * H_ + h] = f2bf(y);
      }
    }
  }
}

// ---------------- 1x1 conv (512x256 GEMM) + GLU + residual ----------------
__global__ __launch_bounds__(256) void mix_glu_kernel(
    const unsigned short* __restrict__ w16L,    // [2H][H] bf16, this layer
    const unsigned short* __restrict__ ygt,     // [B][LYP][H] bf16
    const float* __restrict__ out_bL,           // [2H]
    const float* __restrict__ hA,               // [B][H][L] residual in
    float* __restrict__ hB)                     // [B][H][L] out
{
  __shared__ unsigned short wa[16 * H_];        // 'a' rows  g = hh0..hh0+15
  __shared__ unsigned short wb[16 * H_];        // 'b' rows  g = 256+hh0..
  const int hh0  = blockIdx.x * 16;
  const int tid  = threadIdx.x;
  const int wave = tid >> 5;
  const int lane = tid & 31;

#if HAVE_TDM
  if (wave == 0) {
    tdm_load_2d_b16((unsigned)(size_t)(void*)wa, w16L + (size_t)hh0 * H_, H_, 16, H_);
    tdm_load_2d_b16((unsigned)(size_t)(void*)wb, w16L + (size_t)(H_ + hh0) * H_, H_, 16, H_);
    wait_tensor0();
  }
  __syncthreads();
#else
  for (int i = tid; i < 16 * H_ / 8; i += 256) {
    ((u32x4*)wa)[i] = ((const u32x4*)(w16L + (size_t)hh0 * H_))[i];
    ((u32x4*)wb)[i] = ((const u32x4*)(w16L + (size_t)(H_ + hh0) * H_))[i];
  }
  __syncthreads();
#endif

  const int r      = lane & 15;
  const int kb0A   = (lane < 16) ? 0 : 8;
  const int kb0B   = (lane < 16) ? 0 : 8;
  const int rowadd = (lane < 16) ? 0 : 8;
  const int NT     = B_ * NC;                   // 2504 N-tiles of 16 cols

  for (int t = blockIdx.y * 8 + wave; t < NT; t += gridDim.y * 8) {
    const int bb = t / NC;
    const int l0 = (t - bb * NC) * 16;
    v8f acca = {0.f, 0.f, 0.f, 0.f, 0.f, 0.f, 0.f, 0.f};
    v8f accb = {0.f, 0.f, 0.f, 0.f, 0.f, 0.f, 0.f, 0.f};
    for (int k0 = 0; k0 < H_; k0 += 32) {
      Frag16 fa, fb, fB;
      const u32x4* pa = (const u32x4*)(wa + r * H_ + k0 + kb0A);
      fa.q[0] = pa[0];  fa.q[1] = pa[2];        // +16 elems = +2 u32x4
      const u32x4* pb = (const u32x4*)(wb + r * H_ + k0 + kb0A);
      fb.q[0] = pb[0];  fb.q[1] = pb[2];
      const u32x4* pB = (const u32x4*)(ygt + ((size_t)bb * LYP + l0 + r) * H_ + k0 + kb0B);
      fB.q[0] = pB[0];  fB.q[1] = pB[2];
      acca = __builtin_amdgcn_wmma_f32_16x16x32_bf16(false, fa.v, false, fB.v,
                                                     (short)0, acca, false, false);
      accb = __builtin_amdgcn_wmma_f32_16x16x32_bf16(false, fb.v, false, fB.v,
                                                     (short)0, accb, false, false);
    }
    const int l = l0 + r;
    if (l < L_) {
#pragma unroll
      for (int v = 0; v < 8; ++v) {
        const int g  = hh0 + v + rowadd;
        float av = acca[v] + out_bL[g];
        float bv = accb[v] + out_bL[H_ + g];
        float s  = 1.0f / (1.0f + expf(-bv));   // GLU
        const size_t off = ((size_t)bb * H_ + g) * L_ + l;
        hB[off] = av * s + hA[off];             // residual
      }
    }
  }
}

// ---------------- BatchNorm (training-mode forward) ----------------
__global__ __launch_bounds__(256) void bn_stats_kernel(const float* __restrict__ hB,
                                                       float* __restrict__ stats) {
  __shared__ float ss[256], ss2[256];
  const int h = blockIdx.x, tid = threadIdx.x;
  float s = 0.f, s2 = 0.f;
  for (int t = tid; t < B_ * L_; t += 256) {
    int b = t / L_, l = t - b * L_;
    float v = hB[((size_t)b * H_ + h) * L_ + l];
    s += v;  s2 += v * v;
  }
  ss[tid] = s;  ss2[tid] = s2;  __syncthreads();
  for (int o = 128; o > 0; o >>= 1) {
    if (tid < o) { ss[tid] += ss[tid + o]; ss2[tid] += ss2[tid + o]; }
    __syncthreads();
  }
  if (tid == 0) {
    float n = (float)(B_ * L_);
    float mean = ss[0] / n;
    float var  = ss2[0] / n - mean * mean;
    stats[2 * h]     = mean;
    stats[2 * h + 1] = rsqrtf(var + EPSBN);
  }
}

__global__ void bn_apply_kernel(float* __restrict__ hB, const float* __restrict__ stats,
                                const float* __restrict__ gamma, const float* __restrict__ beta) {
  size_t idx = (size_t)blockIdx.x * 256 + threadIdx.x;
  if (idx >= (size_t)B_ * H_ * L_) return;
  int h = (int)((idx / L_) % H_);
  hB[idx] = (hB[idx] - stats[2 * h]) * stats[2 * h + 1] * gamma[h] + beta[h];
}

__global__ void out_kernel(const float* __restrict__ hA, float* __restrict__ out) {
  size_t idx = (size_t)blockIdx.x * 256 + threadIdx.x;
  if (idx >= (size_t)B_ * L_ * H_) return;
  int h = (int)(idx % H_);
  int l = (int)((idx / H_) % L_);
  int b = (int)(idx / ((size_t)H_ * L_));
  out[idx] = hA[((size_t)b * H_ + h) * L_ + l];
}

// ---------------- host orchestration ----------------
extern "C" void kernel_launch(void* const* d_in, const int* in_sizes, int n_in,
                              void* d_out, int out_size, void* d_ws, size_t ws_size,
                              hipStream_t stream) {
  (void)in_sizes; (void)n_in; (void)out_size; (void)ws_size;
  const float* x      = (const float*)d_in[0];
  const float* pre_w  = (const float*)d_in[1];
  const float* pre_b  = (const float*)d_in[2];
  const float* log_dt = (const float*)d_in[3];
  const float* C_re   = (const float*)d_in[4];
  const float* C_im   = (const float*)d_in[5];
  const float* lAr    = (const float*)d_in[6];
  const float* Aim    = (const float*)d_in[7];
  const float* Dv     = (const float*)d_in[8];
  const float* out_w  = (const float*)d_in[9];
  const float* out_b  = (const float*)d_in[10];
  const float* bng    = (const float*)d_in[11];
  const float* bnb    = (const float*)d_in[12];

  char* ws = (char*)d_ws;
  size_t off = 0;
  auto alloc = [&](size_t bytes) -> void* {
    void* p = (void*)(ws + off);
    off += (bytes + 255) & ~(size_t)255;
    return p;
  };
  float* hA             = (float*)alloc((size_t)B_ * H_ * L_ * 4);           // 41 MB
  float* hB             = (float*)alloc((size_t)B_ * H_ * L_ * 4);           // 41 MB
  unsigned short* u16g  = (unsigned short*)alloc((size_t)B_ * H_ * LUP * 2); // 20.6 MB
  unsigned short* kxr16 = (unsigned short*)alloc((size_t)H_ * LKPR * 2);     // 2.6 MB
  unsigned short* ygt   = (unsigned short*)alloc((size_t)B_ * LYP * H_ * 2); // 20.5 MB
  unsigned short* w16   = (unsigned short*)alloc((size_t)NLAYERS * 2 * H_ * H_ * 2);
  float* Ctr            = (float*)alloc((size_t)H_ * N2_ * 4);
  float* Cti            = (float*)alloc((size_t)H_ * N2_ * 4);
  float* dAr            = (float*)alloc((size_t)H_ * N2_ * 4);
  float* dAi            = (float*)alloc((size_t)H_ * N2_ * 4);
  float* stats          = (float*)alloc((size_t)H_ * 2 * 4);

  {
    size_t n = (size_t)B_ * H_ * L_;
    preproj_kernel<<<(unsigned)((n + 255) / 256), 256, 0, stream>>>(x, pre_w, pre_b, hA);
  }
  {
    int n = NLAYERS * 2 * H_ * H_;
    wcvt_kernel<<<(n + 255) / 256, 256, 0, stream>>>(out_w, w16, n);
  }

  int bnidx = 0;
  for (int layer = 0; layer < NLAYERS; ++layer) {
    ctilde_kernel<<<(H_ * N2_ + 255) / 256, 256, 0, stream>>>(
        log_dt + (size_t)layer * H_, C_re + (size_t)layer * H_ * N2_,
        C_im + (size_t)layer * H_ * N2_, lAr + (size_t)layer * H_ * N2_,
        Aim + (size_t)layer * H_ * N2_, Ctr, Cti, dAr, dAi);

    dim3 gk(LKP / 256, H_);
    kbuild_kernel<<<gk, 256, 0, stream>>>(Ctr, Cti, dAr, dAi, kxr16);

    {
      size_t n = (size_t)B_ * H_ * LUP;
      u16_kernel<<<(unsigned)((n + 255) / 256), 256, 0, stream>>>(hA, u16g);
    }

    conv_wmma_kernel<<<H_, 256, (B_ * LUP + 8 * LKPC) * 2, stream>>>(
        u16g, kxr16, hA, Dv + (size_t)layer * H_, ygt);

    dim3 gm(H_ / 16, 32);
    mix_glu_kernel<<<gm, 256, 0, stream>>>(
        w16 + (size_t)layer * 2 * H_ * H_, ygt, out_b + (size_t)layer * 2 * H_, hA, hB);

    if ((layer & 1) == 0) {
      bn_stats_kernel<<<H_, 256, 0, stream>>>(hB, stats);
      size_t n = (size_t)B_ * H_ * L_;
      bn_apply_kernel<<<(unsigned)((n + 255) / 256), 256, 0, stream>>>(
          hB, stats, bng + (size_t)bnidx * H_, bnb + (size_t)bnidx * H_);
      ++bnidx;
    }
    float* tmp = hA; hA = hB; hB = tmp;
  }

  {
    size_t n = (size_t)B_ * L_ * H_;
    out_kernel<<<(unsigned)((n + 255) / 256), 256, 0, stream>>>(hA, (float*)d_out);
  }
}
